// MoE_73366631350439
// MI455X (gfx1250) — compile-verified, matrix operand order checked by reading, not run on hardware
//
#include <hip/hip_runtime.h>
#include <hip/hip_bf16.h>
#include <math.h>

// ---------------------------------------------------------------------------
// MoE (B=4, T=2048, C=1024, H=4096, E=8, top-2) for gfx1250 (MI455X).
// bf16 WMMA (v_wmma_f32_16x16x32_bf16) grouped-GEMM with fused GELU.
// ---------------------------------------------------------------------------

#define NTOK   8192      // B*T
#define CDIM   1024
#define HDIM   4096
#define NEXP   8

typedef float v8f  __attribute__((ext_vector_type(8)));
typedef __bf16 bf16;
typedef bf16 v16bf __attribute__((ext_vector_type(16)));
typedef bf16 v8bf  __attribute__((ext_vector_type(8)));
typedef bf16 v4bf  __attribute__((ext_vector_type(4)));
typedef float v4f  __attribute__((ext_vector_type(4)));

union BF16V { v16bf v; v8bf h[2]; };

// ---------------- workspace layout (bytes) ----------------
// W1 swizzled bf16 : E*C*H*2      = 64 MiB   @ 0
// W2 swizzled bf16 : E*H*C*2      = 64 MiB   @ 64 MiB
// x bf16           : NTOK*C*2     = 16 MiB   @ 128 MiB
// token lists      : E*NTOK*4     = 256 KiB
// gate lists       : E*NTOK*4     = 256 KiB
// full gate matrix : NTOK*E*4     = 256 KiB
// expert counts    : E*4
static constexpr size_t OFF_W1S  = 0;
static constexpr size_t OFF_W2S  = OFF_W1S + (size_t)NEXP * CDIM * HDIM * 2;
static constexpr size_t OFF_XB   = OFF_W2S + (size_t)NEXP * HDIM * CDIM * 2;
static constexpr size_t OFF_TOK  = OFF_XB + (size_t)NTOK * CDIM * 2;
static constexpr size_t OFF_GAT  = OFF_TOK + (size_t)NEXP * NTOK * 4;
static constexpr size_t OFF_GBUF = OFF_GAT + (size_t)NEXP * NTOK * 4;
static constexpr size_t OFF_CNT  = OFF_GBUF + (size_t)NTOK * NEXP * 4;

// ---------------- tiny init: zero expert counters ----------------
__global__ void moe_init_k(int* cnt) {
    if (threadIdx.x < NEXP) cnt[threadIdx.x] = 0;
}

// ---------------- zero output y (8M floats) ----------------
__global__ void moe_zero_y_k(float* y) {
    size_t i = ((size_t)blockIdx.x * blockDim.x + threadIdx.x) * 4;
    if (i < (size_t)NTOK * CDIM) *(v4f*)(y + i) = (v4f)0.0f;
}

// ---------------- x: f32 -> bf16 ----------------
__global__ void moe_conv_x_k(const float* __restrict__ x, bf16* __restrict__ xb) {
    size_t i = ((size_t)blockIdx.x * blockDim.x + threadIdx.x) * 4;
    if (i >= (size_t)NTOK * CDIM) return;
    v4f s = *(const v4f*)(x + i);
    v4bf d;
    d.x = (bf16)s.x; d.y = (bf16)s.y; d.z = (bf16)s.z; d.w = (bf16)s.w;
    *(v4bf*)(xb + i) = d;
}

// B-operand swizzle (matches 16-bit WMMA B layout, 32x16 per K-slab):
//   within a 512-element (1 KiB) block:
//     pos = load*256 + lane*8 + j ;  k = load*16 + ((lane&16)?8:0) + j ; col = lane&15
// So the FFN kernel loads B with two coalesced 16B-per-lane global_load_b128.
__global__ void moe_swz_w1_k(const float* __restrict__ w1, bf16* __restrict__ out) {
    size_t tid = (size_t)blockIdx.x * blockDim.x + threadIdx.x;
    if (tid >= (size_t)NEXP * CDIM * HDIM) return;
    size_t blk = tid >> 9;            // 1 KiB blocks
    int pos  = (int)(tid & 511);
    int e    = (int)(blk >> 13);      // 256*32 = 8192 blocks / expert
    int t    = (int)((blk >> 5) & 255);   // H tile (16 cols)
    int kc   = (int)(blk & 31);           // C chunk (32 rows)
    int load = pos >> 8;
    int lane = (pos >> 3) & 31;
    int j    = pos & 7;
    int k    = load * 16 + ((lane & 16) ? 8 : 0) + j;
    int col  = lane & 15;
    int c    = kc * 32 + k;
    int h    = t * 16 + col;
    out[tid] = (bf16)w1[((size_t)e * CDIM + c) * HDIM + h];
}

__global__ void moe_swz_w2_k(const float* __restrict__ w2, bf16* __restrict__ out) {
    size_t tid = (size_t)blockIdx.x * blockDim.x + threadIdx.x;
    if (tid >= (size_t)NEXP * HDIM * CDIM) return;
    size_t blk = tid >> 9;
    int pos  = (int)(tid & 511);
    int e    = (int)(blk >> 13);      // 64*128 = 8192 blocks / expert
    int t    = (int)((blk >> 7) & 63);    // C tile (16 cols)
    int kc   = (int)(blk & 127);          // H chunk (32 rows)
    int load = pos >> 8;
    int lane = (pos >> 3) & 31;
    int j    = pos & 7;
    int k    = load * 16 + ((lane & 16) ? 8 : 0) + j;
    int col  = lane & 15;
    int h    = kc * 32 + k;
    int c    = t * 16 + col;
    out[tid] = (bf16)w2[((size_t)e * HDIM + h) * CDIM + c];
}

// ---------------- router: logits -> softmax -> top-2 scatter ----------------
__global__ __launch_bounds__(256)
void moe_router_k(const float* __restrict__ x, const float* __restrict__ wg,
                  const float* __restrict__ rb, int* __restrict__ cnt,
                  int* __restrict__ tok, float* __restrict__ gat,
                  float* __restrict__ gbuf) {
    int lane = threadIdx.x & 31;
    int wave = threadIdx.x >> 5;
    int t = blockIdx.x * 8 + wave;           // 1024 blocks * 8 waves
    if (t >= NTOK) return;
    const float* xr = x + (size_t)t * CDIM;
    float s[NEXP];
#pragma unroll
    for (int e = 0; e < NEXP; ++e) s[e] = 0.f;
    for (int i = 0; i < CDIM / 32; ++i) {
        float xv = xr[lane + 32 * i];
#pragma unroll
        for (int e = 0; e < NEXP; ++e) s[e] += xv * wg[e * CDIM + lane + 32 * i];
    }
#pragma unroll
    for (int off = 16; off > 0; off >>= 1)
#pragma unroll
        for (int e = 0; e < NEXP; ++e) s[e] += __shfl_xor(s[e], off, 32);
    if (lane == 0) {
        float m = -1e30f;
#pragma unroll
        for (int e = 0; e < NEXP; ++e) { s[e] += rb[e]; m = fmaxf(m, s[e]); }
        float den = 0.f, g[NEXP];
#pragma unroll
        for (int e = 0; e < NEXP; ++e) { g[e] = expf(s[e] - m); den += g[e]; }
        float inv = 1.f / den;
#pragma unroll
        for (int e = 0; e < NEXP; ++e) { g[e] *= inv; gbuf[(size_t)t * NEXP + e] = g[e]; }
        int i1 = 0;
#pragma unroll
        for (int e = 1; e < NEXP; ++e) if (g[e] > g[i1]) i1 = e;
        int i2 = (i1 == 0) ? 1 : 0;
#pragma unroll
        for (int e = 0; e < NEXP; ++e) if (e != i1 && g[e] > g[i2]) i2 = e;
        int p1 = atomicAdd(&cnt[i1], 1);
        tok[(size_t)i1 * NTOK + p1] = t; gat[(size_t)i1 * NTOK + p1] = g[i1];
        int p2 = atomicAdd(&cnt[i2], 1);
        tok[(size_t)i2 * NTOK + p2] = t; gat[(size_t)i2 * NTOK + p2] = g[i2];
    }
}

// ---------------- aux loss: fixed-order (deterministic) reduction ----------------
__global__ __launch_bounds__(256)
void moe_aux_k(const float* __restrict__ gbuf, float* __restrict__ out) {
    __shared__ float sh[256 * NEXP];
    int tid = threadIdx.x;
    float s[NEXP];
#pragma unroll
    for (int e = 0; e < NEXP; ++e) s[e] = 0.f;
    for (int t = tid; t < NTOK; t += 256)
#pragma unroll
        for (int e = 0; e < NEXP; ++e) s[e] += gbuf[(size_t)t * NEXP + e];
#pragma unroll
    for (int e = 0; e < NEXP; ++e) sh[tid * NEXP + e] = s[e];
    __syncthreads();
    for (int off = 128; off > 0; off >>= 1) {
        if (tid < off)
#pragma unroll
            for (int e = 0; e < NEXP; ++e) sh[tid * NEXP + e] += sh[(tid + off) * NEXP + e];
        __syncthreads();
    }
    if (tid == 0) {
        float a = 0.f;
#pragma unroll
        for (int e = 0; e < NEXP; ++e) {
            float mi = sh[e] / (float)NTOK;
            a += mi * mi;
        }
        out[(size_t)NTOK * CDIM] = (float)NEXP * a;
    }
}

// ---------------- fused expert FFN: fc1 -> GELU -> fc2, WMMA bf16 ----------------
#define XS_STRIDE 1040     // 1024 + 16 bf16 pad (bank-conflict skew), 16B aligned
#define HS_STRIDE 80       // 64 + 16 pad

__device__ __forceinline__ v16bf load_A16(const bf16* base, int row0, int stride,
                                          int k0, int lane) {
    // 16-bit A layout: lanes 0-15 row m, K=[k0..k0+7] + [k0+16..k0+23];
    //                  lanes 16-31 row m, K shifted by +8.
    int row = row0 + (lane & 15);
    int ko  = k0 + ((lane & 16) ? 8 : 0);
    const bf16* p = base + row * stride + ko;
    BF16V u;
    u.h[0] = *(const v8bf*)(p);
    u.h[1] = *(const v8bf*)(p + 16);
    return u.v;
}

__device__ __forceinline__ v16bf load_B16(const bf16* blk, int lane) {
    // pre-swizzled: two fully-coalesced 16B/lane loads (global_load_b128)
    BF16V u;
    u.h[0] = *(const v8bf*)(blk + lane * 8);
    u.h[1] = *(const v8bf*)(blk + 256 + lane * 8);
    return u.v;
}

__device__ __forceinline__ float gelu_exact(float f) {
    return 0.5f * f * (1.f + erff(f * 0.70710678118654752f));
}

__global__ __launch_bounds__(512)
void moe_ffn_k(const bf16* __restrict__ xb, const bf16* __restrict__ w1s,
               const bf16* __restrict__ w2s, const float* __restrict__ b1,
               const float* __restrict__ b2, const int* __restrict__ cnt,
               const int* __restrict__ tok, const float* __restrict__ gat,
               float* __restrict__ yout) {
    __shared__ bf16 xs[64 * XS_STRIDE];   // 130 KiB token tile (CDNA5: 320 KiB/WGP)
    __shared__ bf16 hs[64 * HS_STRIDE];   // 10 KiB hidden chunk
    __shared__ float gs[64];
    __shared__ int   tts[64];

    int e  = blockIdx.x >> 7;            // 128 token-blocks per expert
    int mb = blockIdx.x & 127;
    int n_tok = cnt[e];
    if (mb * 64 >= n_tok) return;        // block-uniform early exit

    int tid = threadIdx.x;
    int lane = tid & 31;
    int w = tid >> 5;                    // 16 waves

    if (tid < 64) {
        int r = mb * 64 + tid;
        bool valid = r < n_tok;
        tts[tid] = valid ? tok[(size_t)e * NTOK + r] : 0;
        gs[tid]  = valid ? gat[(size_t)e * NTOK + r] : 0.f;   // pads contribute 0
    }
    __syncthreads();

    // cooperative gather of 64 token rows (bf16) into LDS
    for (int idx = tid; idx < 64 * 128; idx += 512) {
        int row = idx >> 7, ch = idx & 127;
        *(v8bf*)(xs + row * XS_STRIDE + ch * 8) =
            *(const v8bf*)(xb + (size_t)tts[row] * CDIM + ch * 8);
    }
    __syncthreads();

    int m_t   = w & 3;                   // fc1: this wave's token-tile
    int n_sub = w >> 2;                  // fc1: this wave's 16 h-columns in chunk
    int n_base = w * 64;                 // fc2: this wave's 64 output columns

    v8f y[4][4];
#pragma unroll
    for (int a = 0; a < 4; ++a)
#pragma unroll
        for (int b = 0; b < 4; ++b) y[a][b] = 0.f;

    const bf16* w1e = w1s + (size_t)e * (256 * 32 * 512);
    const bf16* w2e = w2s + (size_t)e * (64 * 128 * 512);
    int rbase1 = m_t * 16 + ((lane & 16) ? 8 : 0);
    int cl1    = n_sub * 16 + (lane & 15);

    for (int hc = 0; hc < HDIM / 64; ++hc) {
        // ---- fc1: h_tile[16x16] = x_tile[16x1024] @ W1[1024x16] ----
        v8f acc = 0.f;
        const bf16* w1blk = w1e + (size_t)(hc * 4 + n_sub) * (32 * 512);
#pragma unroll 4
        for (int kc = 0; kc < 32; ++kc) {
            v16bf A = load_A16(xs, m_t * 16, XS_STRIDE, kc * 32, lane);
            v16bf B = load_B16(w1blk + kc * 512, lane);
            acc = __builtin_amdgcn_wmma_f32_16x16x32_bf16(
                false, A, false, B, (short)0, acc, false, false);
        }
        // bias + exact GELU, write bf16 h chunk (C/D layout -> row-major LDS)
        float b1v = b1[e * HDIM + hc * 64 + cl1 - n_sub * 16 + n_sub * 16]; // h col
#pragma unroll
        for (int v = 0; v < 8; ++v) {
            float f = acc[v] + b1v;
            hs[(rbase1 + v) * HS_STRIDE + cl1] = (bf16)gelu_exact(f);
        }
        __syncthreads();

        // ---- fc2: y[64 x my64cols] += h[64x64] @ W2[64 x my64cols] ----
#pragma unroll
        for (int mt = 0; mt < 4; ++mt) {
#pragma unroll
            for (int kk = 0; kk < 2; ++kk) {
                v16bf A = load_A16(hs, mt * 16, HS_STRIDE, kk * 32, lane);
#pragma unroll
                for (int nt = 0; nt < 4; ++nt) {
                    const bf16* blk =
                        w2e + (size_t)((w * 4 + nt) * 128 + (hc * 2 + kk)) * 512;
                    v16bf B = load_B16(blk, lane);
                    y[mt][nt] = __builtin_amdgcn_wmma_f32_16x16x32_bf16(
                        false, A, false, B, (short)0, y[mt][nt], false, false);
                }
            }
        }
        __syncthreads();   // protect hs before next chunk overwrites it
    }

    // ---- epilogue: + fc2 bias, * gate, scatter-add (exactly 2 adds/element) ----
#pragma unroll
    for (int mt = 0; mt < 4; ++mt) {
        int rb = mt * 16 + ((lane & 16) ? 8 : 0);
#pragma unroll
        for (int nt = 0; nt < 4; ++nt) {
            int col = n_base + nt * 16 + (lane & 15);
            float b2v = b2[e * CDIM + col];
#pragma unroll
            for (int v = 0; v < 8; ++v) {
                float val = (y[mt][nt][v] + b2v) * gs[rb + v];
                atomicAdd(yout + (size_t)tts[rb + v] * CDIM + col, val);
            }
        }
    }
}

// ---------------------------------------------------------------------------
extern "C" void kernel_launch(void* const* d_in, const int* in_sizes, int n_in,
                              void* d_out, int out_size, void* d_ws, size_t ws_size,
                              hipStream_t stream) {
    const float* x   = (const float*)d_in[0];
    const float* wg  = (const float*)d_in[1];
    const float* rb  = (const float*)d_in[2];
    const float* w1  = (const float*)d_in[3];
    const float* b1  = (const float*)d_in[4];
    const float* w2  = (const float*)d_in[5];
    const float* b2  = (const float*)d_in[6];
    float* y = (float*)d_out;            // [NTOK*CDIM] y  +  [1] aux

    char* ws = (char*)d_ws;
    bf16*  w1s  = (bf16*)(ws + OFF_W1S);
    bf16*  w2s  = (bf16*)(ws + OFF_W2S);
    bf16*  xb   = (bf16*)(ws + OFF_XB);
    int*   tokl = (int*)(ws + OFF_TOK);
    float* gatl = (float*)(ws + OFF_GAT);
    float* gbuf = (float*)(ws + OFF_GBUF);
    int*   cnt  = (int*)(ws + OFF_CNT);

    moe_init_k<<<1, 32, 0, stream>>>(cnt);
    moe_zero_y_k<<<(NTOK * CDIM / 4 + 255) / 256, 256, 0, stream>>>(y);
    moe_conv_x_k<<<(NTOK * CDIM / 4 + 255) / 256, 256, 0, stream>>>(x, xb);
    {
        size_t n = (size_t)NEXP * CDIM * HDIM;
        moe_swz_w1_k<<<(unsigned)((n + 255) / 256), 256, 0, stream>>>(w1, w1s);
        moe_swz_w2_k<<<(unsigned)((n + 255) / 256), 256, 0, stream>>>(w2, w2s);
    }
    moe_router_k<<<NTOK / 8, 256, 0, stream>>>(x, wg, rb, cnt, tokl, gatl, gbuf);
    moe_aux_k<<<1, 256, 0, stream>>>(gbuf, y);
    moe_ffn_k<<<NEXP * (NTOK / 64), 512, 0, stream>>>(xb, w1s, w2s, b1, b2,
                                                      cnt, tokl, gatl, y);
}
